// TestModel_20315195310823
// MI455X (gfx1250) — 8-bit matrix operand order repaired and checked by reading, compiled, not run on hardware
//
#include <hip/hip_runtime.h>
#include <hip/hip_bf16.h>
#include <stdint.h>

constexpr int H = 4096;
constexpr int M = 8192;
constexpr float FP8_MAX = 448.0f;
constexpr float EPS = 1e-6f;

constexpr size_t OFF_W1Q   = 0;
constexpr size_t OFF_W2Q   = (size_t)H * H;
constexpr size_t OFF_ACT   = (size_t)2 * H * H;
constexpr size_t OFF_RESID = OFF_ACT + (size_t)M * H;

typedef __attribute__((ext_vector_type(16))) int          v16i;
typedef __attribute__((ext_vector_type(8)))  float        v8f;
typedef __attribute__((ext_vector_type(8)))  int          v8i;
typedef __attribute__((ext_vector_type(4)))  int          v4i;
typedef __attribute__((ext_vector_type(4)))  unsigned int v4u;
typedef __attribute__((ext_vector_type(2)))  int          v2i;

#if defined(__has_builtin)
#  if __has_builtin(__builtin_amdgcn_tensor_load_to_lds)
#    define HAVE_TDM 1
#  endif
#endif
#ifndef HAVE_TDM
#  define HAVE_TDM 0
#endif

__device__ __forceinline__ unsigned int f32_to_e4m3(float x) {
    x = fminf(FP8_MAX, fmaxf(-FP8_MAX, x));
    unsigned u = __float_as_uint(x);
    unsigned s = (u >> 24) & 0x80u;
    int      e = (int)((u >> 23) & 0xFFu);
    unsigned m = u & 0x7FFFFFu;
    if (e == 0) return s;
    int ue = e - 127;
    if (ue < -9) return s;
    if (ue < -6) {
        int shift = -6 - ue;
        unsigned mm   = 0x800000u | m;
        int      sh   = 21 + shift;
        unsigned q    = mm >> sh;
        unsigned rem  = mm & ((1u << sh) - 1u);
        unsigned half = 1u << (sh - 1);
        if (rem > half || (rem == half && (q & 1u))) q++;
        return s | q;
    }
    unsigned q   = m >> 20;
    unsigned rem = m & 0xFFFFFu;
    if (rem > 0x80000u || (rem == 0x80000u && (q & 1u))) q++;
    unsigned code = (((unsigned)(ue + 7)) << 3) + q;
    if (code > 0x7Eu) code = 0x7Eu;
    return s | code;
}

__global__ __launch_bounds__(256) void quant_w(const float* __restrict__ w,
                                               unsigned char* __restrict__ wq) {
    unsigned id = blockIdx.x * 256u + threadIdx.x;
    unsigned k = id >> 12;
    unsigned n = id & (H - 1);
    wq[(size_t)n * H + k] = (unsigned char)f32_to_e4m3(w[id]);
}

template <int MODE>
__global__ __launch_bounds__(256) void rms_quant(const float* __restrict__ xin,
                                                 const float* __restrict__ g,
                                                 const float* __restrict__ s,
                                                 unsigned int* __restrict__ actq,
                                                 float* __restrict__ resid,
                                                 float* __restrict__ outf) {
    const int row = blockIdx.x;
    const int tid = threadIdx.x;
    const float4* x4 = (const float4*)(xin + (size_t)row * H);
    const float4* g4 = (const float4*)g;

    float4 r[4];
    float ss = 0.f;
#pragma unroll
    for (int c = 0; c < 4; ++c) {
        r[c] = x4[c * 256 + tid];
        ss += r[c].x * r[c].x + r[c].y * r[c].y + r[c].z * r[c].z + r[c].w * r[c].w;
    }
#pragma unroll
    for (int off = 16; off > 0; off >>= 1) ss += __shfl_xor(ss, off, 32);

    __shared__ float red[8];
    __shared__ float rstd_sh;
    if ((tid & 31) == 0) red[tid >> 5] = ss;
    __syncthreads();
    if (tid == 0) {
        float t = 0.f;
#pragma unroll
        for (int i = 0; i < 8; ++i) t += red[i];
        rstd_sh = rsqrtf(t * (1.0f / H) + EPS);
    }
    __syncthreads();
    const float rstd  = rstd_sh;
    const float inv_s = (MODE == 2) ? 1.0f : (1.0f / s[0]);

#pragma unroll
    for (int c = 0; c < 4; ++c) {
        const int idx = c * 256 + tid;
        float4 gg = g4[idx];
        float4 y;
        y.x = r[c].x * rstd * gg.x;
        y.y = r[c].y * rstd * gg.y;
        y.z = r[c].z * rstd * gg.z;
        y.w = r[c].w * rstd * gg.w;
        if (MODE == 2) {
            ((float4*)outf)[(size_t)row * (H / 4) + idx] = y;
        } else {
            unsigned p = f32_to_e4m3(y.x * inv_s)
                       | (f32_to_e4m3(y.y * inv_s) << 8)
                       | (f32_to_e4m3(y.z * inv_s) << 16)
                       | (f32_to_e4m3(y.w * inv_s) << 24);
            actq[(size_t)row * (H / 4) + idx] = p;
            if (MODE == 0) {
                float4 rl;
                rl.x = fmaxf(r[c].x, 0.f); rl.y = fmaxf(r[c].y, 0.f);
                rl.z = fmaxf(r[c].z, 0.f); rl.w = fmaxf(r[c].w, 0.f);
                ((float4*)resid)[(size_t)row * (H / 4) + idx] = rl;
            }
        }
    }
}

#if HAVE_TDM
__device__ __forceinline__ void tdm_load_tile_128x128(uint32_t lds_off,
                                                      const unsigned char* gptr,
                                                      uint32_t stride_elems) {
    uint64_t ga = (uint64_t)(uintptr_t)gptr;
    v4u g0 = { 1u,
               lds_off,
               (uint32_t)ga,
               ((uint32_t)(ga >> 32) & 0x01FFFFFFu) | 0x80000000u };
    v8i g1;
    g1[0] = 0;
    g1[1] = (int)(stride_elems << 16);
    g1[2] = (int)(stride_elems >> 16);
    g1[3] = (int)((128u << 16) | 0x10u);
    g1[4] = 128;
    g1[5] = (int)stride_elems;
    g1[6] = 0;
    g1[7] = 0;
    v4i z4 = { 0, 0, 0, 0 };
#if __clang_major__ >= 23
    v8i z8 = {};
    __builtin_amdgcn_tensor_load_to_lds(g0, g1, z4, z4, z8, 0);
#else
    __builtin_amdgcn_tensor_load_to_lds(g0, g1, z4, z4, 0);
#endif
}
#endif

__global__ __launch_bounds__(256, 1) void gemm_fp8(const unsigned char* __restrict__ Aq,
                                                   const unsigned char* __restrict__ Bq,
                                                   float* __restrict__ resid,
                                                   const float* __restrict__ sa,
                                                   const float* __restrict__ sb) {
    const int tid  = threadIdx.x;
    const int lane = tid & 31;
    const int wave = tid >> 5;
    const int wr   = wave & 1;
    const int wc   = wave >> 1;
    const int m0   = blockIdx.y * 128;
    const int n0   = blockIdx.x * 128;

    const float scale = sa[0] * sb[0];
    v8f acc[4][2] = {};

#if HAVE_TDM
    __shared__ unsigned char ldsA[2][128 * 128];
    __shared__ unsigned char ldsB[2][128 * 128];
    const uint32_t offA0 = (uint32_t)(uintptr_t)&ldsA[0][0];
    const uint32_t offA1 = (uint32_t)(uintptr_t)&ldsA[1][0];
    const uint32_t offB0 = (uint32_t)(uintptr_t)&ldsB[0][0];
    const uint32_t offB1 = (uint32_t)(uintptr_t)&ldsB[1][0];

    if (wave == 0) {
        tdm_load_tile_128x128(offA0, Aq + (size_t)m0 * H, H);
        tdm_load_tile_128x128(offB0, Bq + (size_t)n0 * H, H);
        __builtin_amdgcn_s_wait_tensorcnt(0);
    }
    __syncthreads();
#else
    __shared__ unsigned char ldsA[1][128 * 128];
    __shared__ unsigned char ldsB[1][128 * 128];
#endif

    for (int k0 = 0; k0 < H; k0 += 128) {
#if HAVE_TDM
        const int cur = (k0 >> 7) & 1;
        if (wave == 0 && (k0 + 128) < H) {
            tdm_load_tile_128x128(cur ? offA0 : offA1, Aq + (size_t)m0 * H + k0 + 128, H);
            tdm_load_tile_128x128(cur ? offB0 : offB1, Bq + (size_t)n0 * H + k0 + 128, H);
        }
        const unsigned char* sA = ldsA[cur];
        const unsigned char* sB = ldsB[cur];
#else
        const int cur = 0;
#pragma unroll
        for (int p = 0; p < 4; ++p) {
            const int idx  = p * 256 + tid;
            const int row  = idx >> 3;
            const int c16  = (idx & 7) << 4;
            *(uint4*)&ldsA[0][row * 128 + c16] =
                *(const uint4*)&Aq[(size_t)(m0 + row) * H + k0 + c16];
            *(uint4*)&ldsB[0][row * 128 + c16] =
                *(const uint4*)&Bq[(size_t)(n0 + row) * H + k0 + c16];
        }
        if (k0 + 128 < H) {
            __builtin_prefetch(&Aq[(size_t)(m0 + (tid >> 1)) * H + k0 + 128], 0, 1);
            __builtin_prefetch(&Bq[(size_t)(n0 + (tid >> 1)) * H + k0 + 128], 0, 1);
        }
        __syncthreads();
        const unsigned char* sA = ldsA[0];
        const unsigned char* sB = ldsB[0];
#endif
        (void)cur;

        union { v16i v; v4i q[4]; } bf[2];
#pragma unroll
        for (int nt = 0; nt < 2; ++nt) {
            const int n    = wc * 32 + nt * 16 + (lane & 15);
            const int base = n * 128 + ((lane >> 4) << 4);
#pragma unroll
            for (int i = 0; i < 4; ++i)
                bf[nt].q[i] = *(const v4i*)&sB[base + 32 * i];
        }

#pragma unroll
        for (int mt = 0; mt < 4; ++mt) {
            union { v16i v; v2i h[8]; } af;
            const int mrow = wr * 64 + mt * 16 + (lane & 15);
            const int base = mrow * 128 + ((lane >> 4) << 4);
#pragma unroll
            for (int i = 0; i < 8; ++i)
                af.h[i] = *(const v2i*)&sA[base + 32 * (i >> 1) + 8 * (i & 1)];

#pragma unroll
            for (int nt = 0; nt < 2; ++nt) {
                acc[mt][nt] = __builtin_amdgcn_wmma_f32_16x16x128_fp8_fp8(
                    af.v, bf[nt].v, (short)0, acc[mt][nt], false, false);
            }
        }

#if HAVE_TDM
        if (wave == 0) __builtin_amdgcn_s_wait_tensorcnt(0);
        __syncthreads();
#else
        __syncthreads();
#endif
    }

    const int halfsel = (lane >> 4) << 3;
    const int colbase = n0 + wc * 32 + (lane & 15);
#pragma unroll
    for (int mt = 0; mt < 4; ++mt) {
#pragma unroll
        for (int nt = 0; nt < 2; ++nt) {
#pragma unroll
            for (int j = 0; j < 8; ++j) {
                const int rr = m0 + wr * 64 + mt * 16 + halfsel + j;
                const size_t idx = (size_t)rr * H + colbase + nt * 16;
                resid[idx] = acc[mt][nt][j] * scale + resid[idx];
            }
        }
    }
}

extern "C" void kernel_launch(void* const* d_in, const int* in_sizes, int n_in,
                              void* d_out, int out_size, void* d_ws, size_t ws_size,
                              hipStream_t stream) {
    (void)in_sizes; (void)n_in; (void)out_size; (void)ws_size;
    const float* x  = (const float*)d_in[0];
    const float* w1 = (const float*)d_in[1];
    const float* w2 = (const float*)d_in[2];
    const float* s0 = (const float*)d_in[3];
    const float* s1 = (const float*)d_in[4];
    const float* s2 = (const float*)d_in[5];
    const float* s3 = (const float*)d_in[6];
    const float* g0 = (const float*)d_in[7];
    const float* g1 = (const float*)d_in[8];
    const float* g2 = (const float*)d_in[9];
    float* out = (float*)d_out;

    uint8_t*       ws    = (uint8_t*)d_ws;
    unsigned char* w1q   = ws + OFF_W1Q;
    unsigned char* w2q   = ws + OFF_W2Q;
    unsigned int*  actq  = (unsigned int*)(ws + OFF_ACT);
    float*         resid = (float*)(ws + OFF_RESID);

    const dim3 blk(256);
    const dim3 grid_qw((H * H) / 256);
    const dim3 grid_gemm(H / 128, M / 128);

    quant_w<<<grid_qw, blk, 0, stream>>>(w1, w1q);
    quant_w<<<grid_qw, blk, 0, stream>>>(w2, w2q);

    rms_quant<0><<<M, blk, 0, stream>>>(x, g0, s0, actq, resid, nullptr);
    gemm_fp8<<<grid_gemm, blk, 0, stream>>>((const unsigned char*)actq, w1q, resid, s0, s1);
    rms_quant<1><<<M, blk, 0, stream>>>(resid, g1, s2, actq, nullptr, nullptr);
    gemm_fp8<<<grid_gemm, blk, 0, stream>>>((const unsigned char*)actq, w2q, resid, s2, s3);
    rms_quant<2><<<M, blk, 0, stream>>>(resid, g2, nullptr, nullptr, nullptr, out);
}
